// CrossAttention_33990371180653
// MI455X (gfx1250) — compile-verified, matrix operand order checked by reading, not run on hardware
//
#include <hip/hip_runtime.h>
#include <hip/hip_bf16.h>
#include <stdint.h>

// ---------------------------------------------------------------------------
// CDNA5 (gfx1250) cross-attention:  out = softmax((xWq)(yWk)^T / 32) (yWv) + x
// All GEMM-like ops on v_wmma_f32_16x16x32_bf16 (f32 accumulate).
// K/V double-buffered in LDS via GLOBAL_LOAD_ASYNC_TO_LDS_B128 (ASYNCcnt),
// pipelined one tile ahead of the WMMA/softmax body.
// ---------------------------------------------------------------------------

typedef __attribute__((ext_vector_type(16))) __bf16 v16bf;
typedef __attribute__((ext_vector_type(2)))  __bf16 v2bf;
typedef __attribute__((ext_vector_type(8)))  float  v8f;

// types matching the async-LDS builtin signature (int vector_size(16))
typedef int v4i __attribute__((vector_size(16)));
typedef __attribute__((address_space(1))) v4i* gv4i_p;
typedef __attribute__((address_space(3))) v4i* lv4i_p;

#define BLOCK 256
#define SKT   32            // kv rows per streamed tile
#define NTILE (2048 / SKT)  // 64 kv tiles
#define APW   20            // async loads issued per wave per tile (4 K + 16 V)

#if defined(__gfx1250__) && __has_builtin(__builtin_amdgcn_global_load_async_to_lds_b128)
#define HAVE_ASYNC_LDS 1
#else
#define HAVE_ASYNC_LDS 0
#endif

#if __has_builtin(__builtin_amdgcn_s_wait_asynccnt)
#define WAIT_ASYNC(n) __builtin_amdgcn_s_wait_asynccnt(n)
#else
#define WAIT_ASYNC(n) asm volatile("s_wait_asynccnt %0" ::"i"(n) : "memory")
#endif

// native f32 -> bf16 (lowers to v_cvt_pk_bf16_f32 on gfx1250)
static __device__ __forceinline__ unsigned short f32_bf16(float f) {
    return __builtin_bit_cast(unsigned short, (__bf16)f);
}
static __device__ __forceinline__ unsigned int pack_bf16x2(float lo, float hi) {
    v2bf t;
    t.x = (__bf16)lo;
    t.y = (__bf16)hi;
    return __builtin_bit_cast(unsigned int, t);
}

// keep fragment-load groups from being sunk into the WMMA sequence
static __device__ __forceinline__ void sched_fence() {
#if __has_builtin(__builtin_amdgcn_sched_barrier)
    __builtin_amdgcn_sched_barrier(0);
#endif
}

// Dword v of a 16-bit A/B WMMA fragment holds the bf16 pair starting at this K
// (ISA 7.12.2: lanes 0-15: v0..3 -> K=0..7, v4..7 -> K=16..23; +8 for lanes 16-31)
static __device__ __forceinline__ int kpair(int v, int half) {
    return 2 * v + ((v >= 4) ? 8 : 0) + 8 * half;
}

union FragA { v16bf v; unsigned int u[8]; };

// ---------------------------------------------------------------------------
// Kernel 1: C_bf16[M x N] = A_f32[M x 1024] * W_f32[1024 x N]
// 128x128 tile per block; 8 waves, each: 16-row strip x 8 n-tiles.
// ---------------------------------------------------------------------------
__global__ __launch_bounds__(BLOCK)
void proj_gemm_bf16(const float* __restrict__ A,
                    const float* __restrict__ W,
                    unsigned short* __restrict__ C,
                    int N) {
    __shared__ unsigned int lds_a[128 * 16];   // [row][kpair]  (k contig)
    __shared__ unsigned int lds_w[128 * 16];   // [n][kpair]    (transposed, k contig)

    const int tid  = threadIdx.x;
    const int wv   = tid >> 5;
    const int lane = tid & 31;
    const int half = lane >> 4;
    const int l15  = lane & 15;

    const int m0 = blockIdx.y * 128;
    const int n0 = blockIdx.x * 128;

    v8f acc[8] = {};

    for (int kt = 0; kt < 32; ++kt) {
        const int k0 = kt * 32;
        __syncthreads();
        // stage A tile 128x32 as bf16 pairs: 2048 dwords, 8 per thread
#pragma unroll
        for (int i = 0; i < 8; ++i) {
            int idx = tid + BLOCK * i;               // pair index 0..2047
            int r = idx >> 4, kp = idx & 15;
            const float* ap = A + (size_t)(m0 + r) * 1024 + k0 + 2 * kp;
            lds_a[idx] = pack_bf16x2(ap[0], ap[1]);
        }
        // stage W tile 32x128 transposed -> [n][kpair]
#pragma unroll
        for (int i = 0; i < 8; ++i) {
            int idx = tid + BLOCK * i;               // 0..2047
            int kp = idx >> 7, n = idx & 127;
            const float* wp = W + (size_t)(k0 + 2 * kp) * N + n0 + n;
            lds_w[n * 16 + kp] = pack_bf16x2(wp[0], wp[N]);
        }
        __syncthreads();

        // A fragment for this wave's 16-row strip
        FragA a;
        const int arow = wv * 16 + l15;
#pragma unroll
        for (int v = 0; v < 8; ++v)
            a.u[v] = lds_a[arow * 16 + (kpair(v, half) >> 1)];

        // preload ALL B fragments, then issue 8 back-to-back WMMAs
        FragA b[8];
#pragma unroll
        for (int nt = 0; nt < 8; ++nt)
#pragma unroll
            for (int v = 0; v < 8; ++v)
                b[nt].u[v] = lds_w[(nt * 16 + l15) * 16 + (kpair(v, half) >> 1)];

        sched_fence();
#pragma unroll
        for (int nt = 0; nt < 8; ++nt)
            acc[nt] = __builtin_amdgcn_wmma_f32_16x16x32_bf16(
                false, a.v, false, b[nt].v, (short)0, acc[nt], false, false);
        sched_fence();
    }

    // store (C/D layout: row = r + 8*half, col = lane&15)
#pragma unroll
    for (int nt = 0; nt < 8; ++nt) {
#pragma unroll
        for (int r = 0; r < 8; ++r) {
            int row = m0 + wv * 16 + r + 8 * half;
            int col = n0 + nt * 16 + l15;
            C[(size_t)row * N + col] = f32_bf16(acc[nt][r]);
        }
    }
}

// ---------------------------------------------------------------------------
// Kernel 2: flash attention + residual.
// Block = 32 q rows of one batch; 8 waves as a 2x4 grid: wave (mr,nc) owns
// q rows [16*mr,16*mr+16) and O columns [256*nc,256*nc+256).
// kv streamed in 32-row tiles, double-buffered in LDS via async DMA.
// ---------------------------------------------------------------------------
__global__ __launch_bounds__(BLOCK)
void flash_attn(const unsigned short* __restrict__ qg,   // [B*2048 x 256] bf16
                const unsigned short* __restrict__ kg,   // [B*2048 x 256] bf16
                const unsigned short* __restrict__ vg,   // [B*2048 x 1024] bf16
                const float* __restrict__ x,             // [B*2048 x 1024] f32
                float* __restrict__ out) {               // [B*2048 x 1024] f32
    extern __shared__ unsigned short smem[];
    unsigned short* lds_k = smem;                       // 2 x [SKT][256]
    unsigned short* lds_v = lds_k + 2 * SKT * 256;      // 2 x [SKT][1024]
    unsigned short* p_ws  = lds_v + 2 * SKT * 1024;     // [wave][16][32]

    const int tid  = threadIdx.x;
    const int wv   = tid >> 5;
    const int lane = tid & 31;
    const int half = lane >> 4;
    const int l15  = lane & 15;
    const int mr   = wv >> 2;                     // 0..1 : row block
    const int nc   = wv & 3;                      // 0..3 : col block

    const int b  = blockIdx.y;
    const int q0 = blockIdx.x * 32;               // q row within batch

    // issue one kv tile's async copies into buffer (tt & 1)
    auto stage = [&](int tt) {
        const size_t kv0 = (size_t)b * 2048 + (size_t)tt * SKT;
        const int buf = tt & 1;
        const uint4* ks = (const uint4*)kg + kv0 * 32;
        const uint4* vs = (const uint4*)vg + kv0 * 128;
        uint4* lk4 = (uint4*)(lds_k + buf * SKT * 256);
        uint4* lv4 = (uint4*)(lds_v + buf * SKT * 1024);
#if HAVE_ASYNC_LDS
#pragma unroll
        for (int i = 0; i < (SKT * 32) / BLOCK; ++i) {
            int idx = tid + BLOCK * i;
            __builtin_amdgcn_global_load_async_to_lds_b128(
                (gv4i_p)(ks + idx), (lv4i_p)(lk4 + idx), 0, 0);
        }
#pragma unroll
        for (int i = 0; i < (SKT * 128) / BLOCK; ++i) {
            int idx = tid + BLOCK * i;
            __builtin_amdgcn_global_load_async_to_lds_b128(
                (gv4i_p)(vs + idx), (lv4i_p)(lv4 + idx), 0, 0);
        }
#else
#pragma unroll
        for (int i = 0; i < (SKT * 32) / BLOCK; ++i)
            lk4[tid + BLOCK * i] = ks[tid + BLOCK * i];
#pragma unroll
        for (int i = 0; i < (SKT * 128) / BLOCK; ++i)
            lv4[tid + BLOCK * i] = vs[tid + BLOCK * i];
#endif
    };

    // --- load Q fragments (16 rows x 256 = 8 K-chunks) once, per wave ---
    FragA qf[8];
    {
        const unsigned int* qp =
            (const unsigned int*)qg + ((size_t)b * 2048 + q0 + 16 * mr + l15) * 128;
#pragma unroll
        for (int c = 0; c < 8; ++c)
#pragma unroll
            for (int v = 0; v < 8; ++v)
                qf[c].u[v] = qp[c * 16 + (kpair(v, half) >> 1)];
    }

    v8f o[16] = {};
    float m[8], l[8];
#pragma unroll
    for (int r = 0; r < 8; ++r) { m[r] = -3.0e38f; l[r] = 0.0f; }

    const float scale = 0.03125f;   // 1/sqrt(OUT_DIM=1024)

    stage(0);                        // prologue: tile 0 in flight

    for (int t = 0; t < NTILE; ++t) {
        // issue next tile into the other buffer (its readers finished last iter)
        if (t + 1 < NTILE) stage(t + 1);
        if (t + 2 < NTILE) {        // L2 warm-up for tile t+2
            const size_t kvp = (size_t)b * 2048 + (size_t)(t + 2) * SKT;
            __builtin_prefetch((const char*)((const uint4*)kg + kvp * 32) + tid * 64, 0, 1);
            __builtin_prefetch((const char*)((const uint4*)vg + kvp * 128) + tid * 256, 0, 1);
        }
#if HAVE_ASYNC_LDS
        if (t + 1 < NTILE) WAIT_ASYNC(APW);   // tile t resident (in-order done)
        else               WAIT_ASYNC(0);
#endif
        __syncthreads();            // all waves see tile t

        const unsigned short* ck = lds_k + (t & 1) * SKT * 256;
        const unsigned short* cv = lds_v + (t & 1) * SKT * 1024;

        // --- S = Q * K^T  (16 x 32) for this wave's row block ---
        v8f s0 = {}, s1 = {};
#pragma unroll
        for (int c = 0; c < 8; ++c) {
            FragA bk0, bk1;
#pragma unroll
            for (int v = 0; v < 8; ++v) {
                int kp = kpair(v, half) >> 1;
                bk0.u[v] = ((const unsigned int*)ck)[(l15)      * 128 + c * 16 + kp];
                bk1.u[v] = ((const unsigned int*)ck)[(16 + l15) * 128 + c * 16 + kp];
            }
            s0 = __builtin_amdgcn_wmma_f32_16x16x32_bf16(
                false, qf[c].v, false, bk0.v, (short)0, s0, false, false);
            s1 = __builtin_amdgcn_wmma_f32_16x16x32_bf16(
                false, qf[c].v, false, bk1.v, (short)0, s1, false, false);
        }

        // --- online softmax (row = r + 8*half; cols striped over 16-lane half) ---
        float alpha[8];
#pragma unroll
        for (int r = 0; r < 8; ++r) {
            float rmax = fmaxf(s0[r], s1[r]);
#pragma unroll
            for (int off = 8; off >= 1; off >>= 1)
                rmax = fmaxf(rmax, __shfl_xor(rmax, off, 32));
            float mnew = fmaxf(m[r], rmax * scale);
            float a  = __expf(m[r] - mnew);
            float p0 = __expf(s0[r] * scale - mnew);
            float p1 = __expf(s1[r] * scale - mnew);
            float ps = p0 + p1;
#pragma unroll
            for (int off = 8; off >= 1; off >>= 1)
                ps += __shfl_xor(ps, off, 32);
            l[r] = l[r] * a + ps;
            m[r] = mnew;
            alpha[r] = a;
            int prow = r + 8 * half;
            p_ws[wv * 512 + prow * 32 + l15]      = f32_bf16(p0);
            p_ws[wv * 512 + prow * 32 + 16 + l15] = f32_bf16(p1);
        }

        // rescale accumulator
#pragma unroll
        for (int nt = 0; nt < 16; ++nt)
#pragma unroll
            for (int r = 0; r < 8; ++r)
                o[nt][r] *= alpha[r];

        // P as A-fragment (16x32), from this wave's private LDS buffer
        FragA pf;
#pragma unroll
        for (int v = 0; v < 8; ++v)
            pf.u[v] = ((const unsigned int*)p_ws)[wv * 256 + l15 * 16 +
                                                  (kpair(v, half) >> 1)];

        // --- O += P * V over this wave's 256 columns (2 groups of 8 tiles) ---
#pragma unroll
        for (int g = 0; g < 2; ++g) {
            FragA bvf[8];
#pragma unroll
            for (int j = 0; j < 8; ++j) {
                int nt  = g * 8 + j;
                int col = nc * 256 + nt * 16 + l15;
#pragma unroll
                for (int v = 0; v < 8; ++v) {
                    int kk = kpair(v, half);
                    unsigned int lo = cv[(kk)     * 1024 + col];
                    unsigned int hi = cv[(kk + 1) * 1024 + col];
                    bvf[j].u[v] = lo | (hi << 16);
                }
            }
            sched_fence();
#pragma unroll
            for (int j = 0; j < 8; ++j)
                o[g * 8 + j] = __builtin_amdgcn_wmma_f32_16x16x32_bf16(
                    false, pf.v, false, bvf[j].v, (short)0, o[g * 8 + j], false, false);
            sched_fence();
        }

        __syncthreads();            // buffer (t&1) free for tile t+2's DMA
    }

    // --- epilogue: normalize, add residual, store f32 ---
    float inv[8];
#pragma unroll
    for (int r = 0; r < 8; ++r) inv[r] = 1.0f / l[r];
#pragma unroll
    for (int nt = 0; nt < 16; ++nt) {
        int col = nc * 256 + nt * 16 + l15;
#pragma unroll
        for (int r = 0; r < 8; ++r) {
            int row = q0 + 16 * mr + r + 8 * half;
            size_t idx = ((size_t)b * 2048 + row) * 1024 + col;
            out[idx] = o[nt][r] * inv[r] + x[idx];
        }
    }
}

// ---------------------------------------------------------------------------
extern "C" void kernel_launch(void* const* d_in, const int* in_sizes, int n_in,
                              void* d_out, int out_size, void* d_ws, size_t ws_size,
                              hipStream_t stream) {
    (void)in_sizes; (void)n_in; (void)out_size; (void)ws_size;
    const float* x  = (const float*)d_in[0];
    const float* y  = (const float*)d_in[1];
    const float* Wq = (const float*)d_in[2];
    const float* Wk = (const float*)d_in[3];
    const float* Wv = (const float*)d_in[4];
    float* out = (float*)d_out;

    const size_t M = 16 * 2048;                 // 32768 rows
    unsigned short* qws = (unsigned short*)d_ws;              // [M x 256]
    unsigned short* kws = qws + M * 256;                      // [M x 256]
    unsigned short* vws = kws + M * 256;                      // [M x 1024]

    // projections: q = x Wq, k = y Wk, v = y Wv
    proj_gemm_bf16<<<dim3(256 / 128, M / 128), BLOCK, 0, stream>>>(x, Wq, qws, 256);
    proj_gemm_bf16<<<dim3(256 / 128, M / 128), BLOCK, 0, stream>>>(y, Wk, kws, 256);
    proj_gemm_bf16<<<dim3(1024 / 128, M / 128), BLOCK, 0, stream>>>(y, Wv, vws, 1024);

    // flash attention + residual (double-buffered K/V: 168 KB of the 320 KB WGP LDS)
    const size_t smem_bytes =
        (size_t)(2 * SKT * 256 + 2 * SKT * 1024 + 8 * 512) * sizeof(unsigned short); // 172032
    hipError_t err = hipFuncSetAttribute((const void*)flash_attn,
                                         hipFuncAttributeMaxDynamicSharedMemorySize,
                                         (int)smem_bytes);
    (void)err;
    flash_attn<<<dim3(2048 / 32, 16), BLOCK, smem_bytes, stream>>>(
        qws, kws, vws, x, out);
}
